// FA_13683765805677
// MI455X (gfx1250) — matrix operand order repaired and checked by reading, not compiled, not run on hardware
//
#include <hip/hip_runtime.h>
#include <hip/hip_bf16.h>

typedef _Float16 half_t;
typedef __attribute__((ext_vector_type(16))) _Float16 v16h;
typedef __attribute__((ext_vector_type(8)))  float    v8f;

// ---------------------------------------------------------------------------
// Network constants (fixed by the reference)
// ---------------------------------------------------------------------------
#define BN   8
#define H0   256
#define W0   256
#define NF   64
#define OFFC 54     // IN_NC*2*3*3

// ---------------------------------------------------------------------------
// WMMA fragment helpers (gfx1250 wave32, 16x16x32 f16 -> f32)
//
// Orientation: GEMM  D[M=out-ch 16][N=pixels 16] += A(W)[16x32] * B(X)[32x16]
//   A (16-bit 16x32): lane l: m=l&15, h=l>>4;  VGPR v holds K pair
//       k0(v) = 2v + 16h                             (contiguous, matches B)
//   B (16-bit 32x16): lane l: n=l&15, h=l>>4;  halfs e -> K = 16h + e
//   C/D f32: lane l: n=l&15; VGPR r -> m = r + 8*(l>>4)
// ---------------------------------------------------------------------------
__device__ inline v16h load_a_frag(const half_t* __restrict__ base, int rowStride, int lane) {
    const int m = lane & 15, h = lane >> 4;
    union { v16h v; int i[8]; } f;
    const half_t* p = base + (size_t)m * rowStride;
#pragma unroll
    for (int vv = 0; vv < 8; ++vv) {
        const int k0 = 2 * vv + 16 * h;
        f.i[vv] = *reinterpret_cast<const int*>(p + k0);
    }
    return f.v;
}

__device__ inline v16h load_b_frag(const half_t* __restrict__ base, int pixStrideHalfs, int lane) {
    const int n = lane & 15, h = lane >> 4;
    const half_t* p = base + (size_t)n * pixStrideHalfs + h * 16;
    union { v16h v; int4 q[2]; } f;
    f.q[0] = *reinterpret_cast<const int4*>(p);
    f.q[1] = *reinterpret_cast<const int4*>(p + 8);
    return f.v;
}

__device__ inline void store_c8(half_t* __restrict__ dst, v8f acc,
                                const float* __restrict__ bias,
                                int oc0, int coutValid, bool doRelu) {
    union { int4 q; half_t h[8]; } s;
#pragma unroll
    for (int r = 0; r < 8; ++r) {
        const int oc = oc0 + r;
        float v = (oc < coutValid) ? (acc[r] + bias[oc]) : 0.f;
        if (doRelu) v = fmaxf(v, 0.f);
        s.h[r] = (half_t)v;
    }
    *reinterpret_cast<int4*>(dst) = s.q;
}

// ---------------------------------------------------------------------------
// Generic 3x3 conv as implicit GEMM with WMMA.
// Activations: NHWC f16 with 2-pixel zero halo:  [B][H+4][W+4][64]
// Weights pre-packed f16: [tap(9)][o(64 pitch)][c(CIN)]
// Optional concat along C: channels [0,64) from xA, [64,128) from xB.
// Block = 256 threads = 8 waves; one block per output row (b, oy);
// each wave owns one 16-pixel tile and all 4 out-channel blocks.
// ---------------------------------------------------------------------------
template<int CIN, int STRIDE, bool RELU>
__global__ __launch_bounds__(256) void conv3x3_wmma(
    const half_t* __restrict__ xA, const half_t* __restrict__ xB,
    const half_t* __restrict__ wf, const float* __restrict__ bias,
    half_t* __restrict__ out,
    int Hin, int Win, int Hout, int Wout, int coutValid)
{
    const int XW = Win + 4, OW = Wout + 4;
    const int b  = blockIdx.x / Hout;
    const int oy = blockIdx.x % Hout;
    const int lane = threadIdx.x & 31;
    const int wave = threadIdx.x >> 5;
    const int n = lane & 15, h = lane >> 4;

    const size_t inPlane = (size_t)(Hin + 4) * XW * 64;
    const half_t* xAb = xA + (size_t)b * inPlane;
    const half_t* xBb = (CIN == 128) ? (xB + (size_t)b * inPlane) : xAb;
    half_t* outb = out + (size_t)b * (Hout + 4) * OW * 64;

    const int ntiles = Wout >> 4;
    const v8f vzero = {0.f,0.f,0.f,0.f,0.f,0.f,0.f,0.f};

    for (int t = wave; t < ntiles; t += 8) {
        const int ox0 = t << 4;
        v8f acc[4] = {vzero, vzero, vzero, vzero};

        // prefetch next tile's center row (gfx1250 global_prefetch_b8)
        if (t + 8 < ntiles) {
            const half_t* pf = xAb + ((size_t)(oy * STRIDE + 2) * XW + ((ox0 + 128) * STRIDE + 2)) * 64;
            __builtin_prefetch(pf, 0, 1);
        }

#pragma unroll
        for (int tap = 0; tap < 9; ++tap) {
            const int ky = tap / 3, kx = tap % 3;
            const int ppy  = oy * STRIDE - 1 + ky + 2;          // padded row, always in range
            const int ppx0 = ox0 * STRIDE - 1 + kx + 2;         // padded col for n=0
#pragma unroll
            for (int kc = 0; kc < CIN / 32; ++kc) {
                const int cb = kc * 32;
                const half_t* src = (CIN == 128 && cb >= 64) ? xBb : xAb;
                const int c0 = cb & 63;
                const half_t* bbase = src + ((size_t)ppy * XW + ppx0) * 64 + c0;
                const v16h bfrag = load_b_frag(bbase, STRIDE * 64, lane);
#pragma unroll
                for (int ob = 0; ob < 4; ++ob) {
                    const half_t* abase = wf + ((size_t)tap * 64 + ob * 16) * CIN + cb;
                    const v16h afrag = load_a_frag(abase, CIN, lane);
                    acc[ob] = __builtin_amdgcn_wmma_f32_16x16x32_f16(
                        false, afrag, false, bfrag, (short)0, acc[ob], false, false);
                }
            }
        }

        half_t* orow = outb + ((size_t)(oy + 2) * OW + (ox0 + n + 2)) * 64;
#pragma unroll
        for (int ob = 0; ob < 4; ++ob) {
            const int oc0 = ob * 16 + 8 * h;
            store_c8(orow + oc0, acc[ob], bias, oc0, coutValid, RELU);
        }
    }
}

// ---------------------------------------------------------------------------
// Transposed conv 4x4, stride 2, pad 1 (out = 2*in), as four parity-split
// 2x2-tap stride-1 convs.  Weights pre-packed with flip+transpose applied:
// wf[tap(ky*4+kx)][o][c],  wf[...] = w[c][o][3-ky][3-kx].
// ---------------------------------------------------------------------------
__global__ __launch_bounds__(256) void convT4x4_wmma(
    const half_t* __restrict__ xA, const half_t* __restrict__ wf,
    const float* __restrict__ bias, half_t* __restrict__ out,
    int Hin, int Win, int Hout, int Wout)
{
    const int XW = Win + 4, OW = Wout + 4;
    const int b  = blockIdx.x / Hout;
    const int oy = blockIdx.x % Hout;
    const int lane = threadIdx.x & 31;
    const int wave = threadIdx.x >> 5;
    const int n = lane & 15, h = lane >> 4;

    const half_t* xAb = xA + (size_t)b * (Hin + 4) * XW * 64;
    half_t* outb = out + (size_t)b * (Hout + 4) * OW * 64;

    const int ky0 = oy & 1;
    const int iy0 = (oy + ky0 - 2) >> 1;       // input row for j=0 (can be -1: halo)
    const int ntiles = Wout >> 4;              // both parities interleaved in t
    const v8f vzero = {0.f,0.f,0.f,0.f,0.f,0.f,0.f,0.f};

    for (int t = wave; t < ntiles; t += 8) {
        const int px = t & 1;                  // output-x parity
        const int tt = t >> 1;                 // tile within parity class
        v8f acc[4] = {vzero, vzero, vzero, vzero};

#pragma unroll
        for (int j = 0; j < 2; ++j) {
#pragma unroll
            for (int i = 0; i < 2; ++i) {
                const int ky  = ky0 + 2 * j;
                const int kx  = px  + 2 * i;
                const int tap = ky * 4 + kx;
                const int ppy  = iy0 + j + 2;
                const int ppx0 = (px - 1) + tt * 16 + i + 2;   // input col for n=0
#pragma unroll
                for (int kc = 0; kc < 2; ++kc) {
                    const half_t* bbase = xAb + ((size_t)ppy * XW + ppx0) * 64 + kc * 32;
                    const v16h bfrag = load_b_frag(bbase, 64, lane);
#pragma unroll
                    for (int ob = 0; ob < 4; ++ob) {
                        const half_t* abase = wf + ((size_t)tap * 64 + ob * 16) * 64 + kc * 32;
                        const v16h afrag = load_a_frag(abase, 64, lane);
                        acc[ob] = __builtin_amdgcn_wmma_f32_16x16x32_f16(
                            false, afrag, false, bfrag, (short)0, acc[ob], false, false);
                    }
                }
            }
        }

        const int ox = px + 2 * (tt * 16 + n);
        half_t* orow = outb + ((size_t)(oy + 2) * OW + (ox + 2)) * 64;
#pragma unroll
        for (int ob = 0; ob < 4; ++ob) {
            const int oc0 = ob * 16 + 8 * h;
            store_c8(orow + oc0, acc[ob], bias, oc0, 64, true);
        }
    }
}

// ---------------------------------------------------------------------------
// First conv: Cin=3 (K=27, too thin for WMMA), f32 NCHW in -> f16 NHWC padded.
// One thread = one pixel x 16-output-channel group.
// ---------------------------------------------------------------------------
__global__ __launch_bounds__(256) void conv_in_direct(
    const float* __restrict__ x, const float* __restrict__ w,
    const float* __restrict__ bias, half_t* __restrict__ out)
{
    const long long idx = (long long)blockIdx.x * blockDim.x + threadIdx.x;
    if (idx >= (long long)BN * H0 * W0 * 4) return;
    const int g = (int)(idx & 3);
    const long long p = idx >> 2;
    const int ox = (int)(p & (W0 - 1));
    const int oy = (int)((p >> 8) & (H0 - 1));
    const int b  = (int)(p >> 16);

    float s[27];
#pragma unroll
    for (int c = 0; c < 3; ++c)
#pragma unroll
        for (int tq = 0; tq < 9; ++tq) {
            const int iy = oy - 1 + tq / 3, ix = ox - 1 + tq % 3;
            const bool ok = (iy >= 0 && iy < H0 && ix >= 0 && ix < W0);
            s[c * 9 + tq] = ok ? x[((size_t)(b * 3 + c) * H0 + iy) * W0 + ix] : 0.f;
        }

    union { int4 q[2]; half_t hh[16]; } pk;
#pragma unroll
    for (int r = 0; r < 16; ++r) {
        const int oc = g * 16 + r;
        float a = bias[oc];
#pragma unroll
        for (int q = 0; q < 27; ++q) a += w[oc * 27 + q] * s[q];
        pk.hh[r] = (half_t)fmaxf(a, 0.f);
    }
    half_t* dst = out + (((size_t)b * (H0 + 4) + oy + 2) * (W0 + 4) + ox + 2) * 64 + g * 16;
    reinterpret_cast<int4*>(dst)[0] = pk.q[0];
    reinterpret_cast<int4*>(dst)[1] = pk.q[1];
}

// ---------------------------------------------------------------------------
// Deformable conv: bilinear gather from original f32 input using f16 offsets,
// then 64x27 GEMV per pixel, ReLU, f32 NCHW output.
// ---------------------------------------------------------------------------
__global__ __launch_bounds__(256) void dcn_gather(
    const float* __restrict__ x, const half_t* __restrict__ off,
    const float* __restrict__ w, const float* __restrict__ bias,
    float* __restrict__ out)
{
    const long long p = (long long)blockIdx.x * blockDim.x + threadIdx.x;
    if (p >= (long long)BN * H0 * W0) return;
    const int ox = (int)(p & (W0 - 1));
    const int oy = (int)((p >> 8) & (H0 - 1));
    const int b  = (int)(p >> 16);

    const half_t* op = off + (((size_t)b * (H0 + 4) + oy + 2) * (W0 + 4) + ox + 2) * 64;

    float samp[27];
#pragma unroll
    for (int c = 0; c < 3; ++c)
#pragma unroll
        for (int tq = 0; tq < 9; ++tq) {
            const int ch = (c * 9 + tq) * 2;
            const float oyf = (float)op[ch], oxf = (float)op[ch + 1];
            const float py = (float)(oy - 1 + tq / 3) + oyf;
            const float pxx = (float)(ox - 1 + tq % 3) + oxf;
            const float y0f = floorf(py), x0f = floorf(pxx);
            const int y0 = (int)y0f, xx0 = (int)x0f;
            const float wy = py - y0f, wx = pxx - x0f;
            const float* plane = x + (size_t)(b * 3 + c) * H0 * W0;
            float v00 = 0.f, v01 = 0.f, v10 = 0.f, v11 = 0.f;
            if (y0 >= 0 && y0 < H0) {
                if (xx0 >= 0 && xx0 < W0)         v00 = plane[(size_t)y0 * W0 + xx0];
                if (xx0 + 1 >= 0 && xx0 + 1 < W0) v01 = plane[(size_t)y0 * W0 + xx0 + 1];
            }
            if (y0 + 1 >= 0 && y0 + 1 < H0) {
                if (xx0 >= 0 && xx0 < W0)         v10 = plane[(size_t)(y0 + 1) * W0 + xx0];
                if (xx0 + 1 >= 0 && xx0 + 1 < W0) v11 = plane[(size_t)(y0 + 1) * W0 + xx0 + 1];
            }
            samp[c * 9 + tq] = v00 * (1.f - wy) * (1.f - wx) + v01 * (1.f - wy) * wx
                             + v10 * wy * (1.f - wx)         + v11 * wy * wx;
        }

#pragma unroll 4
    for (int o = 0; o < 64; ++o) {
        float a = bias[o];
#pragma unroll
        for (int q = 0; q < 27; ++q) a += w[o * 27 + q] * samp[q];
        out[(((size_t)b * 64 + o) * H0 + oy) * W0 + ox] = fmaxf(a, 0.f);
    }
}

// ---------------------------------------------------------------------------
// Weight packing:  f32 [O][C][3][3] -> f16 [tap][o (pitch 64)][c]
// ---------------------------------------------------------------------------
__global__ void prep_conv_w(const float* __restrict__ src, half_t* __restrict__ dst,
                            int cout, int cin)
{
    const int total = 9 * cout * cin;
    const int idx = blockIdx.x * blockDim.x + threadIdx.x;
    if (idx >= total) return;
    const int t = idx / (cout * cin);
    const int r = idx % (cout * cin);
    const int o = r / cin, c = r % cin;
    dst[((size_t)t * 64 + o) * cin + c] = (half_t)src[((size_t)o * cin + c) * 9 + t];
}

// convT weights: f32 [C][O][4][4] -> f16 [tap][o][c] with flip+transpose
__global__ void prep_convT_w(const float* __restrict__ src, half_t* __restrict__ dst)
{
    const int idx = blockIdx.x * blockDim.x + threadIdx.x;
    if (idx >= 16 * 64 * 64) return;
    const int t = idx >> 12, o = (idx >> 6) & 63, c = idx & 63;
    const int ky = t >> 2, kx = t & 3;
    dst[((size_t)t * 64 + o) * 64 + c] =
        (half_t)src[((size_t)c * 64 + o) * 16 + (3 - ky) * 4 + (3 - kx)];
}

__global__ void fill_zero_u32(unsigned int* __restrict__ p, long long nWords)
{
    const long long stride = (long long)gridDim.x * blockDim.x;
    for (long long i = (long long)blockIdx.x * blockDim.x + threadIdx.x; i < nWords; i += stride)
        p[i] = 0u;
}

// ---------------------------------------------------------------------------
// Host launcher
// ---------------------------------------------------------------------------
extern "C" void kernel_launch(void* const* d_in, const int* in_sizes, int n_in,
                              void* d_out, int out_size, void* d_ws, size_t ws_size,
                              hipStream_t stream)
{
    // d_in order follows setup_inputs(): inputs, then params in insertion order
    enum { I_X = 0, I_WIN, I_BIN,
           I_D1W1, I_D1B1, I_D1W2, I_D1B2,
           I_D2W1, I_D2B1, I_D2W2, I_D2B2,
           I_TRW1, I_TRB1, I_TRW2, I_TRB2, I_TRWT, I_TRBT,
           I_U2W, I_U2B, I_U2WT, I_U2BT,
           I_U1W, I_U1B, I_U1WT, I_U1BT,
           I_OW, I_OB, I_OFW, I_OFB, I_DW, I_DB };
    const float* F[31];
    for (int i = 0; i < 31 && i < n_in; ++i) F[i] = (const float*)d_in[i];

    // --- workspace layout (all f16, 256B aligned) ---
    size_t cur = 0;
    auto take = [&](size_t halfs) {
        size_t o = cur; cur += ((halfs * 2 + 255) / 256) * 256; return o;
    };
    auto padA = [](int H, int W) { return (size_t)BN * (H + 4) * (W + 4) * 64; };
    char* ws = (char*)d_ws;

    const size_t o_f0  = take(padA(256, 256));
    const size_t o_d1a = take(padA(128, 128));
    const size_t o_f1  = take(padA(128, 128));
    const size_t o_d2a = take(padA(64, 64));
    const size_t o_f2  = take(padA(64, 64));
    const size_t o_t1  = take(padA(32, 32));
    const size_t o_t2  = take(padA(32, 32));
    const size_t o_t   = take(padA(64, 64));
    const size_t o_u2  = take(padA(64, 64));
    const size_t o_u9  = take(padA(128, 128));
    const size_t o_u1b = take(padA(128, 128));
    const size_t o_U   = take(padA(256, 256));
    const size_t o_ow  = take(padA(256, 256));
    const size_t o_off = take(padA(256, 256));

    const size_t w33 = 9 * 64 * 64, w44 = 16 * 64 * 64, w33c = 9 * 64 * 128;
    const size_t o_wd1a = take(w33), o_wd1b = take(w33);
    const size_t o_wd2a = take(w33), o_wd2b = take(w33);
    const size_t o_wt1  = take(w33), o_wt2  = take(w33);
    const size_t o_wtT  = take(w44);
    const size_t o_wu2  = take(w33c), o_wu2T = take(w44);
    const size_t o_wu1  = take(w33c), o_wu1T = take(w44);
    const size_t o_wo   = take(w33),  o_wof  = take(w33);

    auto HP = [&](size_t o) { return (half_t*)(ws + o); };

    // --- 0) zero workspace (halos, padded out-channels) ---
    {
        const long long nWords = (long long)(cur / 4);
        fill_zero_u32<<<2048, 256, 0, stream>>>((unsigned int*)d_ws, nWords);
    }

    // --- 1) pack weights to f16 [tap][o][c] ---
    auto prep = [&](const float* src, size_t dstOff, int cout, int cin) {
        const int total = 9 * cout * cin;
        prep_conv_w<<<(total + 255) / 256, 256, 0, stream>>>(src, HP(dstOff), cout, cin);
    };
    auto prepT = [&](const float* src, size_t dstOff) {
        prep_convT_w<<<(16 * 64 * 64 + 255) / 256, 256, 0, stream>>>(src, HP(dstOff));
    };
    prep(F[I_D1W1], o_wd1a, 64, 64);  prep(F[I_D1W2], o_wd1b, 64, 64);
    prep(F[I_D2W1], o_wd2a, 64, 64);  prep(F[I_D2W2], o_wd2b, 64, 64);
    prep(F[I_TRW1], o_wt1, 64, 64);   prep(F[I_TRW2], o_wt2, 64, 64);
    prepT(F[I_TRWT], o_wtT);
    prep(F[I_U2W], o_wu2, 64, 128);   prepT(F[I_U2WT], o_wu2T);
    prep(F[I_U1W], o_wu1, 64, 128);   prepT(F[I_U1WT], o_wu1T);
    prep(F[I_OW], o_wo, 64, 64);      prep(F[I_OFW], o_wof, OFFC, 64);

    // --- 2) encoder ---
    conv_in_direct<<<(BN * H0 * W0 * 4) / 256, 256, 0, stream>>>(
        F[I_X], F[I_WIN], F[I_BIN], HP(o_f0));

    conv3x3_wmma<64, 2, true><<<BN * 128, 256, 0, stream>>>(
        HP(o_f0), nullptr, HP(o_wd1a), F[I_D1B1], HP(o_d1a), 256, 256, 128, 128, 64);
    conv3x3_wmma<64, 1, true><<<BN * 128, 256, 0, stream>>>(
        HP(o_d1a), nullptr, HP(o_wd1b), F[I_D1B2], HP(o_f1), 128, 128, 128, 128, 64);

    conv3x3_wmma<64, 2, true><<<BN * 64, 256, 0, stream>>>(
        HP(o_f1), nullptr, HP(o_wd2a), F[I_D2B1], HP(o_d2a), 128, 128, 64, 64, 64);
    conv3x3_wmma<64, 1, true><<<BN * 64, 256, 0, stream>>>(
        HP(o_d2a), nullptr, HP(o_wd2b), F[I_D2B2], HP(o_f2), 64, 64, 64, 64, 64);

    // --- 3) trunk ---
    conv3x3_wmma<64, 2, true><<<BN * 32, 256, 0, stream>>>(
        HP(o_f2), nullptr, HP(o_wt1), F[I_TRB1], HP(o_t1), 64, 64, 32, 32, 64);
    conv3x3_wmma<64, 1, true><<<BN * 32, 256, 0, stream>>>(
        HP(o_t1), nullptr, HP(o_wt2), F[I_TRB2], HP(o_t2), 32, 32, 32, 32, 64);
    convT4x4_wmma<<<BN * 64, 256, 0, stream>>>(
        HP(o_t2), HP(o_wtT), F[I_TRBT], HP(o_t), 32, 32, 64, 64);

    // --- 4) decoder (concat handled by dual-tensor B reads) ---
    conv3x3_wmma<128, 1, true><<<BN * 64, 256, 0, stream>>>(
        HP(o_t), HP(o_f2), HP(o_wu2), F[I_U2B], HP(o_u2), 64, 64, 64, 64, 64);
    convT4x4_wmma<<<BN * 128, 256, 0, stream>>>(
        HP(o_u2), HP(o_wu2T), F[I_U2BT], HP(o_u9), 64, 64, 128, 128);

    conv3x3_wmma<128, 1, true><<<BN * 128, 256, 0, stream>>>(
        HP(o_u9), HP(o_f1), HP(o_wu1), F[I_U1B], HP(o_u1b), 128, 128, 128, 128, 64);
    convT4x4_wmma<<<BN * 256, 256, 0, stream>>>(
        HP(o_u1b), HP(o_wu1T), F[I_U1BT], HP(o_U), 128, 128, 256, 256);

    // --- 5) offset head (no ReLU on final offsets, 54 valid channels) ---
    conv3x3_wmma<64, 1, true><<<BN * 256, 256, 0, stream>>>(
        HP(o_U), nullptr, HP(o_wo), F[I_OB], HP(o_ow), 256, 256, 256, 256, 64);
    conv3x3_wmma<64, 1, false><<<BN * 256, 256, 0, stream>>>(
        HP(o_ow), nullptr, HP(o_wof), F[I_OFB], HP(o_off), 256, 256, 256, 256, OFFC);

    // --- 6) deformable conv -> f32 NCHW output with ReLU ---
    dcn_gather<<<(BN * H0 * W0 + 255) / 256, 256, 0, stream>>>(
        F[I_X], HP(o_off), F[I_DW], F[I_DB], (float*)d_out);

    (void)in_sizes; (void)n_in; (void)out_size; (void)ws_size;
}
